// EdgeVGAE_22110491640016
// MI455X (gfx1250) — matrix operand order repaired and checked by reading, not compiled, not run on hardware
//
#include <hip/hip_runtime.h>

// ---------------- CDNA5 / gfx1250 WMMA types ----------------
typedef __bf16 bf16;
typedef bf16  v16bf __attribute__((ext_vector_type(16)));
typedef bf16  v8bf  __attribute__((ext_vector_type(8)));
typedef float v8f   __attribute__((ext_vector_type(8)));
typedef unsigned int u32x4 __attribute__((ext_vector_type(4)));
typedef int          i32x8 __attribute__((ext_vector_type(8)));
typedef int          i32x4 __attribute__((ext_vector_type(4)));

#define IN_D 64
#define ED_D 32
#define H_D  64
#define L_D  32
#define C_D  10
#define G_D  128
#define SLOPE 0.15f

// Native bf16 convert (gfx1250 has hardware bf16 cvt; avoids 3-op integer RNE).
__device__ __forceinline__ bf16 tobf(float f) { return (bf16)f; }

__device__ __forceinline__ float lrelu(float v) { return v > 0.f ? v : SLOPE * v; }

__device__ __forceinline__ v8f wmma_bf16(v16bf a, v16bf b, v8f c) {
  // D(16x16,f32) = A(16x32,bf16) * B(32x16,bf16) + C
  return __builtin_amdgcn_wmma_f32_16x16x32_bf16(false, a, false, b, (short)0, c,
                                                 false, false);
}

__device__ __forceinline__ v16bf combine16(v8bf lo, v8bf hi) {
  return __builtin_shufflevector(lo, hi, 0, 1, 2, 3, 4, 5, 6, 7,
                                 8, 9, 10, 11, 12, 13, 14, 15);
}

// ---- Tensor Data Mover: 1-D f32 block load global -> LDS (one instruction).
// D# group0: count=1 | lds_addr | global_addr(57b) | type=2.
// D# group1: data_size=4B, tensor_dim0 = tile_dim0 = nelem, tile_dim1 = 1.
__device__ __forceinline__ void tdm_load_1d(unsigned lds_off, const void* g, unsigned nelem) {
  unsigned long long ga = (unsigned long long)(size_t)g;
  u32x4 g0 = { 1u, lds_off, (unsigned)ga,
               (unsigned)((ga >> 32) & 0x01ffffffull) | 0x80000000u };
  i32x8 g1 = { (int)0x20000u,                           // data_size=2 (4B)
               (int)((nelem & 0xffffu) << 16),          // tensor_dim0[15:0]
               (int)(((nelem >> 16) & 0xffffu) | (1u << 16)), // dim0 hi | dim1=1
               (int)((nelem & 0xffffu) << 16),          // tile_dim0 (<=65535)
               1,                                       // tile_dim1 = 1
               (int)nelem,                              // dim0_stride lo32
               0, 0 };
  i32x4 gz = { 0, 0, 0, 0 };
#if __clang_major__ >= 23
  i32x8 gz8 = { 0, 0, 0, 0, 0, 0, 0, 0 };
  __builtin_amdgcn_tensor_load_to_lds(g0, g1, gz, gz, gz8, 0);
#else
  __builtin_amdgcn_tensor_load_to_lds(g0, g1, gz, gz, 0);
#endif
}

// A fragment (16x32 logical K-slice) directly from a global f32 row.
// lane: row m = lane&15, half = lane>>4. j0..7 -> K=kbase+half*8+j,
// j8..15 -> K=kbase+16+half*8+(j-8). Two 32B-contiguous runs -> 4x float4.
__device__ __forceinline__ v16bf afrag_g(const float* __restrict__ row, int kbase, int half) {
  const float4* p0 = (const float4*)(row + kbase + half * 8);
  const float4* p1 = (const float4*)(row + kbase + 16 + half * 8);
  float4 a = p0[0], b = p0[1], c = p1[0], d = p1[1];
  v16bf r;
  r[0] = tobf(a.x); r[1] = tobf(a.y); r[2]  = tobf(a.z); r[3]  = tobf(a.w);
  r[4] = tobf(b.x); r[5] = tobf(b.y); r[6]  = tobf(b.z); r[7]  = tobf(b.w);
  r[8] = tobf(c.x); r[9] = tobf(c.y); r[10] = tobf(c.z); r[11] = tobf(c.w);
  r[12] = tobf(d.x); r[13] = tobf(d.y); r[14] = tobf(d.z); r[15] = tobf(d.w);
  return r;
}

// A fragment from an LDS bf16 row-major [16 x ld] staging tile (16B-aligned).
__device__ __forceinline__ v16bf afrag_lds(const bf16* T, int ld, int m, int kbase, int half) {
  const bf16* p0 = T + m * ld + kbase + half * 8;
  v8bf lo = *(const v8bf*)p0;
  v8bf hi = *(const v8bf*)(p0 + 16);
  return combine16(lo, hi);
}

// B fragment from LDS weights stored column-major: Wt[n*Kdim + k] (32B-aligned).
// lane: col n, half = lane>>4; j0..7 -> K = kbase + half*8 + j, j8..15 -> K = kbase + 16 + half*8 + (j-8).
__device__ __forceinline__ v16bf bfrag_lds(const bf16* Wt, int Kdim, int kbase, int ncol, int half) {
  const bf16* p = Wt + ncol * Kdim + kbase + half * 8;
  v8bf lo = *(const v8bf*)p;
  v8bf hi = *(const v8bf*)(p + 16);
  return combine16(lo, hi);
}

// ---------------- Edge conv: emb = MLP(edge_attr); msg = [x_i|emb]@NW+nb;
// ---------------- scatter-add msg into agg[tgt]. One wave = 16 edges. ------
__global__ __launch_bounds__(256) void edge_conv_wmma(
    const float* __restrict__ nodeF,   // [N,64]
    const int*   __restrict__ eidx,    // [2,E]; tgt = eidx + E
    const float* __restrict__ eattr,   // [E,32]
    const float* __restrict__ ew1, const float* __restrict__ eb1,   // [32,64],[64]
    const float* __restrict__ ew2, const float* __restrict__ eb2,   // [64,64],[64]
    const float* __restrict__ nw,  const float* __restrict__ nb,    // [128,64],[64]
    float* __restrict__ agg,           // [N,64], pre-zeroed
    int E, int N)
{
  __shared__ float sWF[ED_D * H_D + H_D * H_D + 2 * H_D * H_D];  // TDM f32 stage
  __shared__ bf16 sW1[ED_D * H_D];        // col-major [n][k]
  __shared__ bf16 sW2[H_D * H_D];
  __shared__ bf16 sNW[H_D * (2 * H_D)];   // n<64, k<128
  __shared__ float sB1[H_D], sB2[H_D], sNB[H_D];
  __shared__ bf16 sT[8][16 * H_D];        // per-wave 16x64 staging

  // --- stage all three weight matrices via the Tensor Data Mover ---
  if (threadIdx.x < 32) {
    unsigned base = (unsigned)(size_t)sWF;           // low 32 bits = LDS offset
    tdm_load_1d(base,                    ew1, ED_D * H_D);
    tdm_load_1d(base + 2048u * 4u,       ew2, H_D * H_D);
    tdm_load_1d(base + 6144u * 4u,       nw,  2 * H_D * H_D);
    __builtin_amdgcn_s_wait_tensorcnt(0);
  }
  __syncthreads();

  // --- convert f32 weights -> bf16, column-major for B fragments ---
  for (int i = threadIdx.x; i < ED_D * H_D; i += 256) {
    int k = i >> 6, n = i & 63; sW1[n * ED_D + k] = tobf(sWF[i]);
  }
  for (int i = threadIdx.x; i < H_D * H_D; i += 256) {
    int k = i >> 6, n = i & 63; sW2[n * H_D + k] = tobf(sWF[2048 + i]);
  }
  for (int i = threadIdx.x; i < 2 * H_D * H_D; i += 256) {
    int k = i >> 6, n = i & 63; sNW[n * 2 * H_D + k] = tobf(sWF[6144 + i]);
  }
  if (threadIdx.x < H_D) {
    sB1[threadIdx.x] = eb1[threadIdx.x];
    sB2[threadIdx.x] = eb2[threadIdx.x];
    sNB[threadIdx.x] = nb[threadIdx.x];
  }
  __syncthreads();

  const int wave = threadIdx.x >> 5;
  const int lane = threadIdx.x & 31;
  const int half = lane >> 4;
  const int nl   = lane & 15;
  const int e0   = (blockIdx.x * 8 + wave) * 16;
  if (e0 >= E) return;

  bf16* T = sT[wave];
  const int* tgt = eidx + E;

  // ---- edge MLP layer 1: t1 = leaky(eattr @ W1 + b1)  (K = ED = 32) ----
  const float* erow = eattr + (size_t)(e0 + nl) * ED_D;
  if (e0 + 128 < E)  // prefetch next block's tile -> global_prefetch_b8
    __builtin_prefetch(eattr + (size_t)(e0 + 128 + nl) * ED_D, 0, 0);
  v16bf Ae = afrag_g(erow, 0, half);
#pragma unroll
  for (int nt = 0; nt < 4; ++nt) {
    v8f acc = {};
    acc = wmma_bf16(Ae, bfrag_lds(sW1, ED_D, 0, nt * 16 + nl, half), acc);
    float bias = sB1[nt * 16 + nl];
#pragma unroll
    for (int r = 0; r < 8; ++r)
      T[(r + 8 * half) * H_D + nt * 16 + nl] = tobf(lrelu(acc[r] + bias));
  }

  // ---- edge MLP layer 2: emb = t1 @ W2 + b2  (K = 64) ----
  v16bf A0 = afrag_lds(T, H_D, nl, 0, half);
  v16bf A1 = afrag_lds(T, H_D, nl, 32, half);
#pragma unroll
  for (int nt = 0; nt < 4; ++nt) {
    v8f acc = {};
    acc = wmma_bf16(A0, bfrag_lds(sW2, H_D, 0,  nt * 16 + nl, half), acc);
    acc = wmma_bf16(A1, bfrag_lds(sW2, H_D, 32, nt * 16 + nl, half), acc);
    float bias = sB2[nt * 16 + nl];
#pragma unroll
    for (int r = 0; r < 8; ++r)
      T[(r + 8 * half) * H_D + nt * 16 + nl] = tobf(acc[r] + bias);  // emb (no act)
  }

  // ---- node MLP: msg = [x_i | emb] @ NW + nb  (K = 128) ----
  const int trow = tgt[e0 + nl];
  const float* xrow = nodeF + (size_t)trow * (size_t)H_D;
  v16bf Ax0 = afrag_g(xrow, 0, half);
  v16bf Ax1 = afrag_g(xrow, 32, half);
  v16bf Am0 = afrag_lds(T, H_D, nl, 0, half);
  v16bf Am1 = afrag_lds(T, H_D, nl, 32, half);

  int trows[8];
#pragma unroll
  for (int r = 0; r < 8; ++r) trows[r] = __shfl(trow, r + 8 * half, 32);

#pragma unroll
  for (int nt = 0; nt < 4; ++nt) {
    v8f acc = {};
    acc = wmma_bf16(Ax0, bfrag_lds(sNW, 2 * H_D, 0,  nt * 16 + nl, half), acc);
    acc = wmma_bf16(Ax1, bfrag_lds(sNW, 2 * H_D, 32, nt * 16 + nl, half), acc);
    acc = wmma_bf16(Am0, bfrag_lds(sNW, 2 * H_D, 64, nt * 16 + nl, half), acc);
    acc = wmma_bf16(Am1, bfrag_lds(sNW, 2 * H_D, 96, nt * 16 + nl, half), acc);
    float bias = sNB[nt * 16 + nl];
#pragma unroll
    for (int r = 0; r < 8; ++r) {
      float v = acc[r] + bias;
      __hip_atomic_fetch_add(&agg[(size_t)trows[r] * H_D + nt * 16 + nl], v,
                             __ATOMIC_RELAXED, __HIP_MEMORY_SCOPE_AGENT);
    }
  }
}

// ---------------- Heads: mu/logvar = h2 @ W + b, z = mu, pooled += z ------
__global__ __launch_bounds__(256) void head_wmma(
    const float* __restrict__ h2, const int* __restrict__ batch,
    const float* __restrict__ muw, const float* __restrict__ mub,
    const float* __restrict__ lvw, const float* __restrict__ lvb,
    float* __restrict__ z_out, float* __restrict__ mu_out, float* __restrict__ lv_out,
    float* __restrict__ pooled, int N)
{
  __shared__ bf16 sMW[H_D * L_D];   // col-major [n][k], n<32, k<64
  __shared__ bf16 sLW[H_D * L_D];
  __shared__ float sMB[L_D], sLB[L_D];

  for (int i = threadIdx.x; i < H_D * L_D; i += 256) {
    int k = i >> 5, n = i & 31;     // muw/lvw row-major [64][32]
    sMW[n * H_D + k] = tobf(muw[i]);
    sLW[n * H_D + k] = tobf(lvw[i]);
  }
  if (threadIdx.x < L_D) { sMB[threadIdx.x] = mub[threadIdx.x]; sLB[threadIdx.x] = lvb[threadIdx.x]; }
  __syncthreads();

  const int wave = threadIdx.x >> 5;
  const int lane = threadIdx.x & 31;
  const int half = lane >> 4;
  const int nl   = lane & 15;
  const int n0   = (blockIdx.x * 8 + wave) * 16;
  if (n0 >= N) return;

  const float* hrow = h2 + (size_t)(n0 + nl) * H_D;
  v16bf A0 = afrag_g(hrow, 0, half);
  v16bf A1 = afrag_g(hrow, 32, half);

  const int brow = batch[n0 + nl];
  int brows[8];
#pragma unroll
  for (int r = 0; r < 8; ++r) brows[r] = __shfl(brow, r + 8 * half, 32);

#pragma unroll
  for (int nt = 0; nt < 2; ++nt) {
    v8f am = {};
    am = wmma_bf16(A0, bfrag_lds(sMW, H_D, 0,  nt * 16 + nl, half), am);
    am = wmma_bf16(A1, bfrag_lds(sMW, H_D, 32, nt * 16 + nl, half), am);
    v8f al = {};
    al = wmma_bf16(A0, bfrag_lds(sLW, H_D, 0,  nt * 16 + nl, half), al);
    al = wmma_bf16(A1, bfrag_lds(sLW, H_D, 32, nt * 16 + nl, half), al);
    float bm = sMB[nt * 16 + nl], bl = sLB[nt * 16 + nl];
#pragma unroll
    for (int r = 0; r < 8; ++r) {
      int node = n0 + r + 8 * half;
      int col  = nt * 16 + nl;
      float vm = am[r] + bm;
      float vl = al[r] + bl;
      z_out[(size_t)node * L_D + col]  = vm;   // z = mu (eps = 0)
      mu_out[(size_t)node * L_D + col] = vm;
      lv_out[(size_t)node * L_D + col] = vl;
      __hip_atomic_fetch_add(&pooled[(size_t)brows[r] * L_D + col], vm,
                             __ATOMIC_RELAXED, __HIP_MEMORY_SCOPE_AGENT);
    }
  }
}

// ---------------- small helper kernels ----------------
__global__ void zero_f32(float* p, int n) {
  int i = blockIdx.x * blockDim.x + threadIdx.x;
  if (i < n) p[i] = 0.f;
}
__global__ void lrelu_inplace(float* p, int n) {
  int i = blockIdx.x * blockDim.x + threadIdx.x;
  if (i < n) { float v = p[i]; p[i] = v > 0.f ? v : SLOPE * v; }
}
__global__ void count_nodes(const int* __restrict__ batch, float* cnt, int n) {
  int i = blockIdx.x * blockDim.x + threadIdx.x;
  if (i < n)
    __hip_atomic_fetch_add(&cnt[batch[i]], 1.0f, __ATOMIC_RELAXED, __HIP_MEMORY_SCOPE_AGENT);
}
__global__ void logits_kernel(const float* __restrict__ pooled, const float* __restrict__ cnt,
                              const float* __restrict__ clsw, const float* __restrict__ clsb,
                              float* __restrict__ out) {
  int g = threadIdx.x;                    // 128 threads
  float c = cnt[g]; c = c > 1.f ? c : 1.f;
  float p[L_D];
#pragma unroll
  for (int l = 0; l < L_D; ++l) p[l] = pooled[g * L_D + l] / c;
#pragma unroll
  for (int cc = 0; cc < C_D; ++cc) {
    float s = clsb[cc];
#pragma unroll
    for (int l = 0; l < L_D; ++l) s += p[l] * clsw[l * C_D + cc];
    out[g * C_D + cc] = s;
  }
}

extern "C" void kernel_launch(void* const* d_in, const int* in_sizes, int n_in,
                              void* d_out, int out_size, void* d_ws, size_t ws_size,
                              hipStream_t stream) {
  const float* x    = (const float*)d_in[0];
  const int*  eidx  = (const int*)d_in[1];
  const float* eattr = (const float*)d_in[2];
  const int*  batch = (const int*)d_in[3];
  const float* e1w1 = (const float*)d_in[4];  const float* e1b1 = (const float*)d_in[5];
  const float* e1w2 = (const float*)d_in[6];  const float* e1b2 = (const float*)d_in[7];
  const float* n1w  = (const float*)d_in[8];  const float* n1b  = (const float*)d_in[9];
  const float* e2w1 = (const float*)d_in[10]; const float* e2b1 = (const float*)d_in[11];
  const float* e2w2 = (const float*)d_in[12]; const float* e2b2 = (const float*)d_in[13];
  const float* n2w  = (const float*)d_in[14]; const float* n2b  = (const float*)d_in[15];
  const float* muw  = (const float*)d_in[16]; const float* mub  = (const float*)d_in[17];
  const float* lvw  = (const float*)d_in[18]; const float* lvb  = (const float*)d_in[19];
  const float* clsw = (const float*)d_in[20]; const float* clsb = (const float*)d_in[21];

  const int N = in_sizes[0] / IN_D;   // 100000
  const int E = in_sizes[2] / ED_D;   // 1600000

  float* agg1   = (float*)d_ws;
  float* agg2   = agg1 + (size_t)N * H_D;
  float* pooled = agg2 + (size_t)N * H_D;
  float* cnt    = pooled + (size_t)G_D * L_D;

  float* z_out  = (float*)d_out;
  float* mu_out = z_out + (size_t)N * L_D;
  float* lv_out = mu_out + (size_t)N * L_D;
  float* logits = lv_out + (size_t)N * L_D;

  const int ztot = N * H_D * 2 + G_D * L_D + G_D;
  zero_f32<<<(ztot + 255) / 256, 256, 0, stream>>>(agg1, ztot);

  const int eblocks = (E + 127) / 128;   // 8 waves x 16 edges per block
  const int nh = N * H_D;
  edge_conv_wmma<<<eblocks, 256, 0, stream>>>(x, eidx, eattr,
      e1w1, e1b1, e1w2, e1b2, n1w, n1b, agg1, E, N);
  lrelu_inplace<<<(nh + 255) / 256, 256, 0, stream>>>(agg1, nh);
  edge_conv_wmma<<<eblocks, 256, 0, stream>>>(agg1, eidx, eattr,
      e2w1, e2b1, e2w2, e2b2, n2w, n2b, agg2, E, N);
  lrelu_inplace<<<(nh + 255) / 256, 256, 0, stream>>>(agg2, nh);

  const int hblocks = (N + 127) / 128;
  head_wmma<<<hblocks, 256, 0, stream>>>(agg2, batch, muw, mub, lvw, lvb,
      z_out, mu_out, lv_out, pooled, N);
  count_nodes<<<(N + 255) / 256, 256, 0, stream>>>(batch, cnt, N);
  logits_kernel<<<1, G_D, 0, stream>>>(pooled, cnt, clsw, clsb, logits);
}
